// Res_GTN_7911329759779
// MI455X (gfx1250) — matrix operand order repaired and checked by reading, not compiled, not run on hardware
//
#include <hip/hip_runtime.h>

#define DEV __device__ __forceinline__

typedef __attribute__((ext_vector_type(16))) __bf16 v16bf;
typedef __attribute__((ext_vector_type(8)))  __bf16 v8bf;
typedef __attribute__((ext_vector_type(8)))  float  v8f;

constexpr int NN = 50000;   // nodes
constexpr int NE = 800000;  // edges

DEV float blo(unsigned u) { return __uint_as_float(u << 16); }
DEV float bhi(unsigned u) { return __uint_as_float(u & 0xffff0000u); }
DEV unsigned short f2bf(float f) {
  unsigned u = __float_as_uint(f);
  unsigned r = u + 0x7FFFu + ((u >> 16) & 1u);
  return (unsigned short)(r >> 16);
}

// ---------------- utility / pack kernels ----------------

__global__ void k_f32_to_bf16(const float* __restrict__ in,
                              unsigned short* __restrict__ out, int n) {
  int i = blockIdx.x * blockDim.x + threadIdx.x;
  if (i < n) out[i] = f2bf(in[i]);
}

// Pack fp32 W[K,Nc] row-major into WMMA-B fragment order:
// Bp[((k32*Nc + col)*2 + ksel)*16 + j], k = k32*32 + ksel*16 + j
DEV size_t packIdx(int k, int col, int Nc) {
  int k32 = k >> 5, krem = k & 31, ksel = krem >> 4, j = krem & 15;
  return (((size_t)k32 * Nc + col) * 2 + ksel) * 16 + j;
}

__global__ void k_pack_w(const float* __restrict__ W,
                         unsigned short* __restrict__ Bp, int K, int Nc) {
  int i = blockIdx.x * blockDim.x + threadIdx.x;
  if (i >= K * Nc) return;
  int k = i / Nc, col = i - k * Nc;
  Bp[packIdx(k, col, Nc)] = f2bf(W[i]);
}

// Bg (256x256, packed): K index a = h*64+d, col b = h2*64+c
// g = q @ Bg  =>  g[n, h*64+c] = sum_d q[n,h*64+d] * We[c, h*64+d]
__global__ void k_build_Bg(const float* __restrict__ We, unsigned short* __restrict__ Bp) {
  int i = blockIdx.x * blockDim.x + threadIdx.x;
  if (i >= 256 * 256) return;
  int a = i >> 8, b = i & 255;
  int h = a >> 6, d = a & 63, h2 = b >> 6, c = b & 63;
  float v = (h == h2) ? We[c * 256 + h * 64 + d] : 0.f;
  Bp[packIdx(a, b, 256)] = f2bf(v);
}

// Bt (256x64, packed): K index a = h*64+c, col d.  out_e = t @ Bt
__global__ void k_build_Bt(const float* __restrict__ We, unsigned short* __restrict__ Bp) {
  int i = blockIdx.x * blockDim.x + threadIdx.x;
  if (i >= 256 * 64) return;
  int a = i >> 6, d = i & 63;
  int h = a >> 6, c = a & 63;
  Bp[packIdx(a, d, 64)] = f2bf(We[c * 256 + h * 64 + d]);
}

// ---------------- WMMA GEMM ----------------
// C[M,Nc] (+)= scale*A@Bp (+bias).  A bf16 [M,K] row-major; Bp packed.
// Each wave produces a 16x64 strip (4 accumulators share one A fragment).
// M%16==0, K%32==0, Nc%64==0.
template <bool ACCUM, bool HASBIAS, bool OUTF32, bool OUTBF16>
__launch_bounds__(128)
__global__ void k_gemm(const unsigned short* __restrict__ A,
                       const unsigned short* __restrict__ Bp,
                       const float* __restrict__ bias,
                       float* __restrict__ Cf,
                       unsigned short* __restrict__ Cb,
                       int M, int K, int Nc, float scale, int totalWaves) {
  int wave = blockIdx.x * (blockDim.x >> 5) + (threadIdx.x >> 5);
  if (wave >= totalWaves) return;        // wave-uniform: EXEC stays all-ones
  int lane = threadIdx.x & 31;
  int colGroups = Nc >> 6;
  int rowTile = wave / colGroups;
  int colGroup = wave - rowTile * colGroups;
  int mL = lane & 15, ksel = lane >> 4;

  const __bf16* Ab = reinterpret_cast<const __bf16*>(A);
  const __bf16* Bb = reinterpret_cast<const __bf16*>(Bp);
  size_t arow = (size_t)(rowTile * 16 + mL) * (size_t)K;
  int nBase = colGroup * 64;

  v8f acc[4] = {v8f{}, v8f{}, v8f{}, v8f{}};
  for (int kk = 0; kk < K; kk += 32) {
    // A 16x32 fragment: one contiguous 32B run per lane (k = 16*ksel + s)
    const __bf16* ap = Ab + arow + kk + ksel * 16;
    v8bf a0 = *reinterpret_cast<const v8bf*>(ap);
    v8bf a1 = *reinterpret_cast<const v8bf*>(ap + 8);
    v16bf a = __builtin_shufflevector(a0, a1, 0, 1, 2, 3, 4, 5, 6, 7,
                                      8, 9, 10, 11, 12, 13, 14, 15);
    size_t bk = (size_t)(kk >> 5) * (size_t)Nc;
#pragma unroll
    for (int t = 0; t < 4; ++t) {
      // packed B fragment: one contiguous 32B run per lane
      const __bf16* bp = Bb + ((bk + nBase + t * 16 + mL) * 2 + ksel) * 16;
      v16bf b = *reinterpret_cast<const v16bf*>(bp);
      acc[t] = __builtin_amdgcn_wmma_f32_16x16x32_bf16(false, a, false, b,
                                                       (short)0, acc[t], false, false);
    }
  }
  int mbase = rowTile * 16 + ksel * 8;
#pragma unroll
  for (int t = 0; t < 4; ++t) {
    int nCol = nBase + t * 16 + mL;
    float bv = HASBIAS ? bias[nCol] : 0.f;
#pragma unroll
    for (int r = 0; r < 8; ++r) {
      size_t idx = (size_t)(mbase + r) * (size_t)Nc + nCol;
      float v = acc[t][r] * scale + bv;
      if (OUTF32) {
        if (ACCUM) v += Cf[idx];
        Cf[idx] = v;
      }
      if (OUTBF16) Cb[idx] = f2bf(v);
    }
  }
}

// ---------------- edge pass 1: alpha + segment-max ----------------
// alpha[e,h] = ( q[dst,h]·k[src,h] + ea[e]·g[dst,h] ) / sqrt(D)
__launch_bounds__(256)
__global__ void k_edge_alpha(const long long* __restrict__ ei,
                             const float* __restrict__ ea,
                             const unsigned short* __restrict__ qb,
                             const unsigned short* __restrict__ kb,
                             const unsigned short* __restrict__ gb,
                             float* __restrict__ alpha,
                             unsigned int* __restrict__ amax_u) {
  int e = blockIdx.x * blockDim.x + threadIdx.x;
  if (e >= NE) return;
  int s = (int)ei[e];
  int d = (int)ei[(size_t)NE + e];
  const uint4* q4 = reinterpret_cast<const uint4*>(qb + (size_t)d * 256);
  const uint4* k4 = reinterpret_cast<const uint4*>(kb + (size_t)s * 256);
  const uint4* g4 = reinterpret_cast<const uint4*>(gb + (size_t)d * 256);
  const float4* e4 = reinterpret_cast<const float4*>(ea + (size_t)e * 64);
  float er[64];
#pragma unroll
  for (int w = 0; w < 16; ++w) {
    float4 f = e4[w];
    er[w * 4 + 0] = f.x; er[w * 4 + 1] = f.y;
    er[w * 4 + 2] = f.z; er[w * 4 + 3] = f.w;
  }
#pragma unroll
  for (int h = 0; h < 4; ++h) {
    float s1 = 0.f, s2 = 0.f;
#pragma unroll
    for (int w = 0; w < 8; ++w) {  // 8 bf16 per uint4
      uint4 qa = q4[h * 8 + w], ka = k4[h * 8 + w], ga = g4[h * 8 + w];
      s1 += blo(qa.x) * blo(ka.x) + bhi(qa.x) * bhi(ka.x);
      s1 += blo(qa.y) * blo(ka.y) + bhi(qa.y) * bhi(ka.y);
      s1 += blo(qa.z) * blo(ka.z) + bhi(qa.z) * bhi(ka.z);
      s1 += blo(qa.w) * blo(ka.w) + bhi(qa.w) * bhi(ka.w);
      const float* ew = er + w * 8;
      s2 += ew[0] * blo(ga.x) + ew[1] * bhi(ga.x);
      s2 += ew[2] * blo(ga.y) + ew[3] * bhi(ga.y);
      s2 += ew[4] * blo(ga.z) + ew[5] * bhi(ga.z);
      s2 += ew[6] * blo(ga.w) + ew[7] * bhi(ga.w);
    }
    float al = (s1 + s2) * 0.125f;  // 1/sqrt(64)
    alpha[(size_t)e * 4 + h] = al;
    unsigned u = __float_as_uint(al);
    u = (u & 0x80000000u) ? ~u : (u | 0x80000000u);  // order-preserving encode
    atomicMax(&amax_u[d * 4 + h], u);
  }
}

// ---------------- edge pass 2: exp + segment-sum ----------------
__launch_bounds__(256)
__global__ void k_edge_exp(const long long* __restrict__ ei,
                           float* __restrict__ alpha,
                           const unsigned int* __restrict__ amax_u,
                           float* __restrict__ denom) {
  int i = blockIdx.x * blockDim.x + threadIdx.x;
  if (i >= NE * 4) return;
  int e = i >> 2, h = i & 3;
  int d = (int)ei[(size_t)NE + e];
  unsigned u = amax_u[d * 4 + h];
  float m = (u & 0x80000000u) ? __uint_as_float(u & 0x7FFFFFFFu)
                              : __uint_as_float(~u);
  float ex = __expf(alpha[i] - m);
  alpha[i] = ex;
  atomicAdd(&denom[d * 4 + h], ex);
}

// ---------------- edge pass 3: aggregate v-part and t = attn^T @ ea ----------------
__launch_bounds__(256)
__global__ void k_edge_agg(const long long* __restrict__ ei,
                           const float* __restrict__ ea,
                           const unsigned short* __restrict__ vb,
                           const float* __restrict__ alpha,
                           const float* __restrict__ denom,
                           float* __restrict__ outacc,   // [N,64]
                           float* __restrict__ t) {      // [N,256]
  int e = blockIdx.x * blockDim.x + threadIdx.x;
  if (e >= NE) return;
  int s = (int)ei[e];
  int d = (int)ei[(size_t)NE + e];
  float4 al = *reinterpret_cast<const float4*>(alpha + (size_t)e * 4);
  float at[4];
  at[0] = al.x / (denom[d * 4 + 0] + 1e-16f);
  at[1] = al.y / (denom[d * 4 + 1] + 1e-16f);
  at[2] = al.z / (denom[d * 4 + 2] + 1e-16f);
  at[3] = al.w / (denom[d * 4 + 3] + 1e-16f);

  const uint4* v4 = reinterpret_cast<const uint4*>(vb + (size_t)s * 256);
  float* orow = outacc + (size_t)d * 64;
#pragma unroll
  for (int w = 0; w < 8; ++w) {
    uint4 v0 = v4[w], v1 = v4[8 + w], v2 = v4[16 + w], v3 = v4[24 + w];
    float c0 = 0.25f * (at[0] * blo(v0.x) + at[1] * blo(v1.x) + at[2] * blo(v2.x) + at[3] * blo(v3.x));
    float c1 = 0.25f * (at[0] * bhi(v0.x) + at[1] * bhi(v1.x) + at[2] * bhi(v2.x) + at[3] * bhi(v3.x));
    float c2 = 0.25f * (at[0] * blo(v0.y) + at[1] * blo(v1.y) + at[2] * blo(v2.y) + at[3] * blo(v3.y));
    float c3 = 0.25f * (at[0] * bhi(v0.y) + at[1] * bhi(v1.y) + at[2] * bhi(v2.y) + at[3] * bhi(v3.y));
    float c4 = 0.25f * (at[0] * blo(v0.z) + at[1] * blo(v1.z) + at[2] * blo(v2.z) + at[3] * blo(v3.z));
    float c5 = 0.25f * (at[0] * bhi(v0.z) + at[1] * bhi(v1.z) + at[2] * bhi(v2.z) + at[3] * bhi(v3.z));
    float c6 = 0.25f * (at[0] * blo(v0.w) + at[1] * blo(v1.w) + at[2] * blo(v2.w) + at[3] * blo(v3.w));
    float c7 = 0.25f * (at[0] * bhi(v0.w) + at[1] * bhi(v1.w) + at[2] * bhi(v2.w) + at[3] * bhi(v3.w));
    atomicAdd(&orow[w * 8 + 0], c0);
    atomicAdd(&orow[w * 8 + 1], c1);
    atomicAdd(&orow[w * 8 + 2], c2);
    atomicAdd(&orow[w * 8 + 3], c3);
    atomicAdd(&orow[w * 8 + 4], c4);
    atomicAdd(&orow[w * 8 + 5], c5);
    atomicAdd(&orow[w * 8 + 6], c6);
    atomicAdd(&orow[w * 8 + 7], c7);
  }
  float* trow = t + (size_t)d * 256;
  const float4* ef = reinterpret_cast<const float4*>(ea + (size_t)e * 64);
#pragma unroll
  for (int w = 0; w < 16; ++w) {
    float4 ec = ef[w];
#pragma unroll
    for (int h = 0; h < 4; ++h) {
      atomicAdd(&trow[h * 64 + w * 4 + 0], at[h] * ec.x);
      atomicAdd(&trow[h * 64 + w * 4 + 1], at[h] * ec.y);
      atomicAdd(&trow[h * 64 + w * 4 + 2], at[h] * ec.z);
      atomicAdd(&trow[h * 64 + w * 4 + 3], at[h] * ec.w);
    }
  }
}

// ---------------- finalize: GroupNorm(16 groups of 4) + exact GELU ----------------
__launch_bounds__(128)
__global__ void k_finalize(const float* __restrict__ outacc,
                           const float* __restrict__ gw, const float* __restrict__ gb,
                           unsigned short* __restrict__ hb_next,
                           float* __restrict__ final_out, int writeOut) {
  int n = blockIdx.x * blockDim.x + threadIdx.x;
  if (n >= NN) return;
  const float4* row = reinterpret_cast<const float4*>(outacc + (size_t)n * 64);
  unsigned short hb[64];
#pragma unroll
  for (int g = 0; g < 16; ++g) {
    float4 a = row[g];  // one group == one float4
    float mu = 0.25f * (a.x + a.y + a.z + a.w);
    float d0 = a.x - mu, d1 = a.y - mu, d2 = a.z - mu, d3 = a.w - mu;
    float var = 0.25f * (d0 * d0 + d1 * d1 + d2 * d2 + d3 * d3);
    float rs = rsqrtf(var + 1e-5f);
    float y[4] = {d0 * rs, d1 * rs, d2 * rs, d3 * rs};
#pragma unroll
    for (int j = 0; j < 4; ++j) {
      int c = g * 4 + j;
      float z = y[j] * gw[c] + gb[c];
      float gel = 0.5f * z * (1.f + erff(z * 0.70710678118654752f));
      hb[c] = f2bf(gel);
      if (writeOut) final_out[(size_t)n * 64 + c] = gel;
    }
  }
  uint4* dstv = reinterpret_cast<uint4*>(hb_next + (size_t)n * 64);
  const uint4* srcv = reinterpret_cast<const uint4*>(hb);
#pragma unroll
  for (int w = 0; w < 8; ++w) dstv[w] = srcv[w];
}

// ---------------- host launcher ----------------

extern "C" void kernel_launch(void* const* d_in, const int* in_sizes, int n_in,
                              void* d_out, int out_size, void* d_ws, size_t ws_size,
                              hipStream_t stream) {
  (void)in_sizes; (void)n_in; (void)out_size; (void)ws_size;

  const float*     x   = (const float*)d_in[0];
  const long long* ei  = (const long long*)d_in[1];
  const float*     ea  = (const float*)d_in[2];
  const float*     gw  = (const float*)d_in[30];
  const float*     gbb = (const float*)d_in[31];

  char* ws = (char*)d_ws;
  size_t off = 0;
  auto alloc = [&](size_t bytes) -> void* {
    void* p = ws + off;
    off += (bytes + 255) & ~(size_t)255;
    return p;
  };
  unsigned short* q_bf   = (unsigned short*)alloc((size_t)NN * 256 * 2);
  unsigned short* k_bf   = (unsigned short*)alloc((size_t)NN * 256 * 2);
  unsigned short* v_bf   = (unsigned short*)alloc((size_t)NN * 256 * 2);
  unsigned short* g_bf   = (unsigned short*)alloc((size_t)NN * 256 * 2);
  float*          alphaB = (float*)alloc((size_t)NE * 4 * 4);
  unsigned int*   amax_u = (unsigned int*)alloc((size_t)NN * 4 * 4);
  float*          denom  = (float*)alloc((size_t)NN * 4 * 4);
  float*          outacc = (float*)alloc((size_t)NN * 64 * 4);
  float*          t_f32  = (float*)alloc((size_t)NN * 256 * 4);
  unsigned short* t_bf   = (unsigned short*)alloc((size_t)NN * 256 * 2);
  unsigned short* hA     = (unsigned short*)alloc((size_t)NN * 128 * 2);
  unsigned short* hB     = (unsigned short*)alloc((size_t)NN * 64 * 2);
  unsigned short* wq_bf  = (unsigned short*)alloc((size_t)128 * 256 * 2);
  unsigned short* wk_bf  = (unsigned short*)alloc((size_t)128 * 256 * 2);
  unsigned short* wv_bf  = (unsigned short*)alloc((size_t)128 * 256 * 2);
  unsigned short* Bg_bf  = (unsigned short*)alloc((size_t)256 * 256 * 2);
  unsigned short* Bt_bf  = (unsigned short*)alloc((size_t)256 * 64 * 2);
  unsigned short* wsk_bf = (unsigned short*)alloc((size_t)128 * 64 * 2);

  // x -> bf16
  k_f32_to_bf16<<<(NN * 128 + 255) / 256, 256, 0, stream>>>(x, hA, NN * 128);
  unsigned short* h_in = hA;

  for (int layer = 0; layer < 3; ++layer) {
    int fin = (layer == 0) ? 128 : 64;
    const float* Wq    = (const float*)d_in[3 + layer * 9 + 0];
    const float* bq    = (const float*)d_in[3 + layer * 9 + 1];
    const float* Wk    = (const float*)d_in[3 + layer * 9 + 2];
    const float* bk    = (const float*)d_in[3 + layer * 9 + 3];
    const float* Wv    = (const float*)d_in[3 + layer * 9 + 4];
    const float* bv    = (const float*)d_in[3 + layer * 9 + 5];
    const float* We    = (const float*)d_in[3 + layer * 9 + 6];
    const float* Wskip = (const float*)d_in[3 + layer * 9 + 7];
    const float* bskip = (const float*)d_in[3 + layer * 9 + 8];
    unsigned short* h_out = (layer & 1) ? hA : hB;

    // weight prep (packed into WMMA B-fragment layout)
    k_pack_w<<<(fin * 256 + 255) / 256, 256, 0, stream>>>(Wq, wq_bf, fin, 256);
    k_pack_w<<<(fin * 256 + 255) / 256, 256, 0, stream>>>(Wk, wk_bf, fin, 256);
    k_pack_w<<<(fin * 256 + 255) / 256, 256, 0, stream>>>(Wv, wv_bf, fin, 256);
    k_pack_w<<<(fin * 64 + 255) / 256, 256, 0, stream>>>(Wskip, wsk_bf, fin, 64);
    k_build_Bg<<<(256 * 256 + 255) / 256, 256, 0, stream>>>(We, Bg_bf);
    k_build_Bt<<<(256 * 64 + 255) / 256, 256, 0, stream>>>(We, Bt_bf);

    hipMemsetAsync(amax_u, 0, (size_t)NN * 4 * 4, stream);
    hipMemsetAsync(denom, 0, (size_t)NN * 4 * 4, stream);
    hipMemsetAsync(outacc, 0, (size_t)NN * 64 * 4, stream);
    hipMemsetAsync(t_f32, 0, (size_t)NN * 256 * 4, stream);

    // node projections (WMMA): bf16 out only
    {
      int waves = (NN / 16) * (256 / 64);
      int blocks = (waves + 3) / 4;
      k_gemm<false, true, false, true><<<blocks, 128, 0, stream>>>(
          h_in, wq_bf, bq, nullptr, q_bf, NN, fin, 256, 1.f, waves);
      k_gemm<false, true, false, true><<<blocks, 128, 0, stream>>>(
          h_in, wk_bf, bk, nullptr, k_bf, NN, fin, 256, 1.f, waves);
      k_gemm<false, true, false, true><<<blocks, 128, 0, stream>>>(
          h_in, wv_bf, bv, nullptr, v_bf, NN, fin, 256, 1.f, waves);
      // g = q @ Bg (factored edge-key projection), no bias
      k_gemm<false, false, false, true><<<blocks, 128, 0, stream>>>(
          q_bf, Bg_bf, nullptr, nullptr, g_bf, NN, 256, 256, 1.f, waves);
    }

    // edge passes (L2-resident gathers + streamed edge_attr)
    k_edge_alpha<<<(NE + 255) / 256, 256, 0, stream>>>(ei, ea, q_bf, k_bf, g_bf,
                                                       alphaB, amax_u);
    k_edge_exp<<<(NE * 4 + 255) / 256, 256, 0, stream>>>(ei, alphaB, amax_u, denom);
    k_edge_agg<<<(NE + 255) / 256, 256, 0, stream>>>(ei, ea, v_bf, alphaB, denom,
                                                     outacc, t_f32);

    // out += 0.25 * t @ Bt (factored edge-value projection); out += skip
    k_f32_to_bf16<<<(NN * 256 + 255) / 256, 256, 0, stream>>>(t_f32, t_bf, NN * 256);
    {
      int waves = (NN / 16) * (64 / 64);
      int blocks = (waves + 3) / 4;
      k_gemm<true, false, true, false><<<blocks, 128, 0, stream>>>(
          t_bf, Bt_bf, nullptr, outacc, nullptr, NN, 256, 64, 0.25f, waves);
      k_gemm<true, true, true, false><<<blocks, 128, 0, stream>>>(
          h_in, wsk_bf, bskip, outacc, nullptr, NN, fin, 64, 1.f, waves);
    }

    k_finalize<<<(NN + 127) / 128, 128, 0, stream>>>(outacc, gw, gbb, h_out,
                                                     (float*)d_out, layer == 2 ? 1 : 0);
    h_in = h_out;
  }
}
